// multi_head_attention_72146860638921
// MI455X (gfx1250) — compile-verified, matrix operand order checked by reading, not run on hardware
//
#include <hip/hip_runtime.h>
#include <stdint.h>

typedef __bf16 bf16_t;
typedef __bf16 v16bf __attribute__((ext_vector_type(16)));
typedef __bf16 bf8v  __attribute__((ext_vector_type(8)));
typedef __bf16 bf4v  __attribute__((ext_vector_type(4)));
typedef float  v8f   __attribute__((ext_vector_type(8)));
typedef float  f4v   __attribute__((ext_vector_type(4)));

#define N_AG 8
#define B_SZ 4096
#define D_SZ 1024
#define H_SZ 8
#define HD_SZ 128
#define M_SZ (N_AG * B_SZ)     /* 32768 */
#define NCOL (H_SZ * HD_SZ)    /* 1024  */

// ---------------------------------------------------------------------------
// f32 -> bf16 elementwise convert (4 elements / thread)
// ---------------------------------------------------------------------------
__global__ __launch_bounds__(256) void cvt_f32_bf16(const float* __restrict__ src,
                                                    bf16_t* __restrict__ dst, int n4) {
    int i = blockIdx.x * 256 + threadIdx.x;
    if (i >= n4) return;
    f4v v = *(const f4v*)(src + 4 * (size_t)i);
    bf4v o;
    o[0] = (bf16_t)v[0]; o[1] = (bf16_t)v[1];
    o[2] = (bf16_t)v[2]; o[3] = (bf16_t)v[3];
    *(bf4v*)(dst + 4 * (size_t)i) = o;
}

// ---------------------------------------------------------------------------
// W[h][d][e] (f32) -> Wt[h*128+e][d] (bf16): K-contiguous rows for B fragments
// ---------------------------------------------------------------------------
__global__ __launch_bounds__(256) void transpose_w(const float* __restrict__ W,
                                                   bf16_t* __restrict__ Wt) {
    int u   = blockIdx.x * 256 + threadIdx.x;   // 1024 blocks -> 262144 threads
    int row = u >> 8;                           // 0..1023  == h*128+e
    int k   = (u & 255) * 4;                    // 4 consecutive d per thread
    int h = row >> 7, e = row & 127;
    bf4v o;
#pragma unroll
    for (int i = 0; i < 4; ++i)
        o[i] = (bf16_t)W[((size_t)h * D_SZ + (k + i)) * HD_SZ + e];
    *(bf4v*)(Wt + (size_t)row * D_SZ + k) = o;
}

// ---------------------------------------------------------------------------
// Build a 16x32 bf16 WMMA A/B fragment from a K-contiguous LDS row chunk.
// lane L: row/col = L%16, half = L/16;  K(e) = 16*(e/8) + 8*half + (e%8)
// -> two contiguous 16B chunks at half*8 and 16+half*8.
// ---------------------------------------------------------------------------
__device__ inline v16bf load_frag(const bf16_t* base, int half) {
    bf8v lo = *(const bf8v*)(base + half * 8);
    bf8v hi = *(const bf8v*)(base + 16 + half * 8);
    v16bf r;
#pragma unroll
    for (int t = 0; t < 8; ++t) { r[t] = lo[t]; r[t + 8] = hi[t]; }
    return r;
}

// ---------------------------------------------------------------------------
// C[M=32768, 1024] = A[M,1024] x Wt^T  (Wt stored [n][k], bf16)
// MODE 0: store bf16 (Q, K).   MODE 1: store f32 with +bias, leaky_relu (V).
// Block: 256 thr / 8 waves, tile 128x128, BK=64 double-buffered LDS.
// Wave (wm,wn) computes 32M x 64N via 8 accumulator fragments.
// ---------------------------------------------------------------------------
template <int MODE>
__global__ __launch_bounds__(256) void gemm_proj(const bf16_t* __restrict__ A,
                                                 const bf16_t* __restrict__ Bt,
                                                 bf16_t* __restrict__ Cbf,
                                                 float* __restrict__ Cf,
                                                 const float* __restrict__ bias) {
    constexpr int LDA = 72;   // padded row stride (halves): 36*r mod 64 distinct
    constexpr int BK  = 64;
    constexpr int KT  = D_SZ / BK;  // 16 stages
    __shared__ __align__(16) bf16_t ldsA[2][128 * LDA];
    __shared__ __align__(16) bf16_t ldsB[2][128 * LDA];

    const int tid  = threadIdx.x;
    const int lane = tid & 31;
    const int wave = tid >> 5;
    const int wm = wave & 3, wn = wave >> 2;
    const int half = lane >> 4, lmod = lane & 15;
    const size_t m0 = (size_t)blockIdx.x * 128;
    const size_t n0 = (size_t)blockIdx.y * 128;

    // branch-free cooperative staging mapping: thread t covers 32 halves (64B)
    // of row (t>>1), half-row (t&1), for both A and B tiles.
    const int srow  = tid >> 1;
    const int scol  = (tid & 1) * 32;
    const bf16_t* gArow = A  + (m0 + srow) * D_SZ + scol;
    const bf16_t* gBrow = Bt + (n0 + srow) * D_SZ + scol;
    bf16_t* sArow = &ldsA[0][srow * LDA + scol];
    bf16_t* sBrow = &ldsB[0][srow * LDA + scol];
    const int lds_buf_stride = 128 * LDA;

    v8f acc[2][4];
#pragma unroll
    for (int mt = 0; mt < 2; ++mt)
#pragma unroll
        for (int nt = 0; nt < 4; ++nt)
#pragma unroll
            for (int v = 0; v < 8; ++v) acc[mt][nt][v] = 0.f;

    // stage 0 into buffer 0
#pragma unroll
    for (int c = 0; c < 32; c += 8) {
        *(uint4*)(sArow + c) = *(const uint4*)(gArow + c);
        *(uint4*)(sBrow + c) = *(const uint4*)(gBrow + c);
    }

    for (int kt = 0; kt < KT; ++kt) {
        const int buf = kt & 1;
        __syncthreads();  // staging of `buf` complete / readers of buf^1 done

        if (kt + 1 < KT) {  // stage next tile into the other buffer
            const int nk = (kt + 1) * BK;
            bf16_t* dA = sArow + (buf ^ 1) * lds_buf_stride;
            bf16_t* dB = sBrow + (buf ^ 1) * lds_buf_stride;
#pragma unroll
            for (int c = 0; c < 32; c += 8) {
                *(uint4*)(dA + c) = *(const uint4*)(gArow + nk + c);
                *(uint4*)(dB + c) = *(const uint4*)(gBrow + nk + c);
            }
        }
        if (kt + 2 < KT) {  // prefetch the tile after that into cache
            const int pk = (kt + 2) * BK;
            __builtin_prefetch(gArow + pk, 0, 0);
            __builtin_prefetch(gBrow + pk, 0, 0);
        }

        // compute on current buffer: 2 K-sub-steps of 32
        const bf16_t* tA = &ldsA[buf][0];
        const bf16_t* tB = &ldsB[buf][0];
#pragma unroll
        for (int ks = 0; ks < 2; ++ks) {
            v16bf af[2], bfr[4];
#pragma unroll
            for (int mt = 0; mt < 2; ++mt)
                af[mt] = load_frag(tA + (wm * 32 + mt * 16 + lmod) * LDA + ks * 32, half);
#pragma unroll
            for (int nt = 0; nt < 4; ++nt)
                bfr[nt] = load_frag(tB + (wn * 64 + nt * 16 + lmod) * LDA + ks * 32, half);
#pragma unroll
            for (int mt = 0; mt < 2; ++mt)
#pragma unroll
                for (int nt = 0; nt < 4; ++nt)
                    acc[mt][nt] = __builtin_amdgcn_wmma_f32_16x16x32_bf16(
                        false, af[mt], false, bfr[nt], (short)0, acc[mt][nt],
                        false, false);
        }
    }

    // epilogue: C/D layout -> lane row m = 8*half + v, col n = lmod
#pragma unroll
    for (int mt = 0; mt < 2; ++mt)
#pragma unroll
        for (int nt = 0; nt < 4; ++nt) {
            size_t colg = n0 + (size_t)wn * 64 + nt * 16 + lmod;
#pragma unroll
            for (int v = 0; v < 8; ++v) {
                size_t rowg = m0 + (size_t)wm * 32 + mt * 16 + half * 8 + v;
                float val = acc[mt][nt][v];
                if (MODE == 0) {
                    Cbf[rowg * NCOL + colg] = (bf16_t)val;
                } else {
                    val += bias[colg];
                    val = val > 0.f ? val : 0.01f * val;  // leaky_relu
                    Cf[rowg * NCOL + colg] = val;
                }
            }
        }
}

// ---------------------------------------------------------------------------
// Leave-one-out attention: one wave per (h,b). Lane l owns dims [4l,4l+4).
// Q,K bf16 / V f32 stored as [n*B+b][h*128+e].  out: [N,H,B,HD].
// Also accumulates unscaled off-diagonal logits -> partial[h*B+b].
// ---------------------------------------------------------------------------
__global__ __launch_bounds__(256) void attention_kernel(
    const bf16_t* __restrict__ Q, const bf16_t* __restrict__ K,
    const float* __restrict__ V, float* __restrict__ out,
    float* __restrict__ partial) {
    const int lane = threadIdx.x & 31;
    const int wave = threadIdx.x >> 5;
    const int gwid = blockIdx.x * 8 + wave;  // 0..H*B-1
    const int h = gwid >> 12;                // / 4096
    const int b = gwid & 4095;
    const int e0 = lane * 4;

    f4v q[8], k[8], v[8];
#pragma unroll
    for (int j = 0; j < 8; ++j) {
        size_t off = (size_t)(j * B_SZ + b) * NCOL + h * HD_SZ + e0;
        bf4v qb = *(const bf4v*)(Q + off);
        bf4v kb = *(const bf4v*)(K + off);
        v[j] = *(const f4v*)(V + off);
#pragma unroll
        for (int t = 0; t < 4; ++t) {
            q[j][t] = (float)qb[t];
            k[j][t] = (float)kb[t];
        }
    }

    const float scale = 0.08838834764831845f;  // 1/sqrt(128)
    float regacc = 0.f;
#pragma unroll
    for (int i = 0; i < 8; ++i) {
        float l[8];
#pragma unroll
        for (int j = 0; j < 8; ++j) {
            f4v p = q[i] * k[j];
            float s = p[0] + p[1] + p[2] + p[3];
#pragma unroll
            for (int off = 16; off > 0; off >>= 1) s += __shfl_xor(s, off, 32);
            l[j] = s;  // wave-uniform
        }
        float m = -__builtin_inff();
#pragma unroll
        for (int j = 0; j < 8; ++j)
            if (j != i) {
                regacc += l[j];
                m = fmaxf(m, l[j] * scale);
            }
        float p[8], s = 0.f;
#pragma unroll
        for (int j = 0; j < 8; ++j) {
            p[j] = (j == i) ? 0.f : __expf(l[j] * scale - m);
            s += p[j];
        }
        float inv = 1.f / s;
        f4v o = {0.f, 0.f, 0.f, 0.f};
#pragma unroll
        for (int j = 0; j < 8; ++j) o += (p[j] * inv) * v[j];
        *(f4v*)(out + ((size_t)(i * H_SZ + h) * B_SZ + b) * HD_SZ + e0) = o;
    }
    if (lane == 0) partial[gwid] = regacc;
}

// ---------------------------------------------------------------------------
// Deterministic final reduction of 32768 partials -> reg_atten scalar
// ---------------------------------------------------------------------------
__global__ __launch_bounds__(256) void reduce_kernel(const float* __restrict__ part,
                                                     float* __restrict__ out) {
    __shared__ float s[256];
    float a = 0.f;
    for (int i = threadIdx.x; i < H_SZ * B_SZ; i += 256) a += part[i];
    s[threadIdx.x] = a;
    __syncthreads();
    for (int o = 128; o > 0; o >>= 1) {
        if ((int)threadIdx.x < o) s[threadIdx.x] += s[threadIdx.x + o];
        __syncthreads();
    }
    if (threadIdx.x == 0)
        out[0] = 0.001f * s[0] / ((float)B_SZ * (float)(N_AG - 1));
}

// ---------------------------------------------------------------------------
extern "C" void kernel_launch(void* const* d_in, const int* in_sizes, int n_in,
                              void* d_out, int out_size, void* d_ws, size_t ws_size,
                              hipStream_t stream) {
    (void)in_sizes; (void)n_in; (void)out_size; (void)ws_size;
    const float* sa = (const float*)d_in[0];  // [N,B,D]
    const float* s  = (const float*)d_in[1];  // [N,B,D]
    const float* WQ = (const float*)d_in[2];  // [H,D,HD]
    const float* WK = (const float*)d_in[3];
    const float* WV = (const float*)d_in[4];
    const float* bV = (const float*)d_in[5];  // [H,HD]
    float* out = (float*)d_out;

    char* ws = (char*)d_ws;
    const size_t MiB = (size_t)1 << 20;
    bf16_t* s_bf  = (bf16_t*)(ws + 0);            // 64 MiB
    bf16_t* sa_bf = (bf16_t*)(ws + 64 * MiB);     // 64 MiB
    bf16_t* wtq   = (bf16_t*)(ws + 128 * MiB);    // 2 MiB
    bf16_t* wtk   = (bf16_t*)(ws + 130 * MiB);    // 2 MiB
    bf16_t* wtv   = (bf16_t*)(ws + 132 * MiB);    // 2 MiB
    bf16_t* Qb    = (bf16_t*)(ws + 134 * MiB);    // 64 MiB
    bf16_t* Kb    = (bf16_t*)(ws + 198 * MiB);    // 64 MiB
    float*  Vf    = (float* )(ws + 262 * MiB);    // 128 MiB
    float*  part  = (float* )(ws + 390 * MiB);    // 128 KiB

    const int n4 = M_SZ * D_SZ / 4;  // 8388608
    cvt_f32_bf16<<<n4 / 256, 256, 0, stream>>>(s, s_bf, n4);
    cvt_f32_bf16<<<n4 / 256, 256, 0, stream>>>(sa, sa_bf, n4);
    transpose_w<<<1024, 256, 0, stream>>>(WQ, wtq);
    transpose_w<<<1024, 256, 0, stream>>>(WK, wtk);
    transpose_w<<<1024, 256, 0, stream>>>(WV, wtv);

    dim3 g(M_SZ / 128, NCOL / 128);  // (256, 8)
    gemm_proj<0><<<g, 256, 0, stream>>>(s_bf,  wtq, Qb, nullptr, nullptr);
    gemm_proj<0><<<g, 256, 0, stream>>>(sa_bf, wtk, Kb, nullptr, nullptr);
    gemm_proj<1><<<g, 256, 0, stream>>>(sa_bf, wtv, nullptr, Vf, bV);

    attention_kernel<<<H_SZ * B_SZ / 8, 256, 0, stream>>>(Qb, Kb, Vf, out, part);
    reduce_kernel<<<1, 256, 0, stream>>>(part, out + (size_t)M_SZ * NCOL);
}
